// _PointnetSAModuleBase_7438883356921
// MI455X (gfx1250) — matrix operand order repaired and checked by reading, not compiled, not run on hardware
//
#include <hip/hip_runtime.h>
#include <hip/hip_bf16.h>

typedef __attribute__((ext_vector_type(16))) _Float16 v16h;
typedef __attribute__((ext_vector_type(8)))  _Float16 v8h;
typedef __attribute__((ext_vector_type(8)))  float    v8f;

#define NB      8
#define NPTS    16384
#define NCH     64
#define NPOINT  1024
#define NSAMPLE 32
#define R2      0.04f
#define GS      40   // LDS row stride in halfs (80B, 16B aligned)

// ================= kernel 1: furthest point sampling ==================
// One 1024-thread block per batch. Each thread keeps its 16 points' coords
// and running min-distance in VGPRs; per step: wave shfl-xor argmax reduce,
// then 32-entry LDS block reduce. Writes new_xyz directly into d_out.
__global__ __launch_bounds__(1024) void fps_kernel(const float* __restrict__ xyz,
                                                   float* __restrict__ new_xyz) {
  const int b   = blockIdx.x;
  const int tid = threadIdx.x;
  __shared__ float red_d[32];
  __shared__ int   red_i[32];
  __shared__ int   sel;

  float px[16], py[16], pz[16], dist[16];
  const float* base = xyz + (size_t)b * NPTS * 3;
#pragma unroll
  for (int i = 0; i < 16; ++i) {
    int p = tid + i * 1024;
    px[i] = base[p * 3 + 0];
    py[i] = base[p * 3 + 1];
    pz[i] = base[p * 3 + 2];
    dist[i] = 1e10f;
  }
  float lx = base[0], ly = base[1], lz = base[2];
  if (tid == 0) {
    float* o = new_xyz + (size_t)b * NPOINT * 3;
    o[0] = lx; o[1] = ly; o[2] = lz;
  }
  const int lane = tid & 31;
  const int wid  = tid >> 5;

  for (int j = 1; j < NPOINT; ++j) {
    float bestd = -1.f;
    int   besti = 0;
#pragma unroll
    for (int i = 0; i < 16; ++i) {
      float dx = px[i] - lx, dy = py[i] - ly, dz = pz[i] - lz;
      float d  = dx * dx + dy * dy + dz * dz;
      float nd = fminf(dist[i], d);
      dist[i]  = nd;
      if (nd > bestd) { bestd = nd; besti = tid + i * 1024; }
    }
#pragma unroll
    for (int off = 16; off > 0; off >>= 1) {
      float od = __shfl_xor(bestd, off);
      int   oi = __shfl_xor(besti, off);
      if (od > bestd || (od == bestd && oi < besti)) { bestd = od; besti = oi; }
    }
    if (lane == 0) { red_d[wid] = bestd; red_i[wid] = besti; }
    __syncthreads();
    if (tid < 32) {
      float bd = red_d[tid];
      int   bi = red_i[tid];
#pragma unroll
      for (int off = 16; off > 0; off >>= 1) {
        float od = __shfl_xor(bd, off);
        int   oi = __shfl_xor(bi, off);
        if (od > bd || (od == bd && oi < bi)) { bd = od; bi = oi; }
      }
      if (tid == 0) sel = bi;
    }
    __syncthreads();
    const int s = sel;
    lx = base[s * 3 + 0]; ly = base[s * 3 + 1]; lz = base[s * 3 + 2];
    if (tid == 0) {
      float* o = new_xyz + ((size_t)b * NPOINT + j) * 3;
      o[0] = lx; o[1] = ly; o[2] = lz;
    }
  }
}

// ================= kernel 2: ball query ==================
// One wave32 per query; collect first NSAMPLE indices (ascending) with
// d2 <= r^2 via ballot + bit peel; pad with the first hit.
__global__ __launch_bounds__(256) void ballq_kernel(const float* __restrict__ xyz,
                                                    const float* __restrict__ new_xyz,
                                                    int* __restrict__ idxbuf) {
  const int gt   = blockIdx.x * 256 + threadIdx.x;
  const int w    = gt >> 5;
  const int lane = gt & 31;
  const int b    = w / NPOINT;
  const int q    = w % NPOINT;
  const float* base = xyz + (size_t)b * NPTS * 3;
  const float* np   = new_xyz + ((size_t)b * NPOINT + q) * 3;
  const float qx = np[0], qy = np[1], qz = np[2];

  int cnt = 0, myidx = -1;
  for (int s = 0; s < NPTS && cnt < NSAMPLE; s += 32) {
    int p = s + lane;
    float dx = base[p * 3 + 0] - qx;
    float dy = base[p * 3 + 1] - qy;
    float dz = base[p * 3 + 2] - qz;
    float d2 = dx * dx + dy * dy + dz * dz;
    unsigned mask = (unsigned)__ballot(d2 <= R2);
    while (mask && cnt < NSAMPLE) {
      int bit = __builtin_ctz(mask);
      if (cnt == lane) myidx = s + bit;
      mask &= mask - 1;
      ++cnt;
    }
  }
  int first = __shfl(myidx, 0);
  if (first < 0) first = 0;
  int out = (myidx >= 0) ? myidx : first;
  idxbuf[((size_t)b * NPOINT + q) * NSAMPLE + lane] = out;
}

// ================= kernel 3: weight prep (f32 -> f16, W1 padded to K=96) ==
__global__ void prep_weights(const float* __restrict__ W1, const float* __restrict__ W2,
                             const float* __restrict__ W3,
                             _Float16* __restrict__ W1h, _Float16* __restrict__ W2h,
                             _Float16* __restrict__ W3h) {
  const int stride = gridDim.x * blockDim.x;
  const int i0 = blockIdx.x * blockDim.x + threadIdx.x;
  for (int t = i0; t < 64 * 96; t += stride) {
    int r = t / 96, k = t % 96;
    W1h[t] = (_Float16)((k < 67) ? W1[r * 67 + k] : 0.f);
  }
  for (int t = i0; t < 64 * 64; t += stride) W2h[t] = (_Float16)W2[t];
  for (int t = i0; t < 128 * 64; t += stride) W3h[t] = (_Float16)W3[t];
}

// ================= WMMA helpers (CDNA5 wave32 layouts) ==================
__device__ inline v8f wmma16(v16h a, v16h b, v8f c) {
  return __builtin_amdgcn_wmma_f32_16x16x32_f16(false, a, false, b, (short)0, c,
                                                false, false);
}

// A 16x32 f16 tile: lane -> m = lane&15, kbase = (lane&16)?8:0.
// elements 0..7 cover k = kbase..kbase+7; elements 8..15 cover k = kbase+16..+23.
__device__ inline v16h load_a(const _Float16* __restrict__ W, int ld, int mt, int kc,
                              int lane) {
  int m  = mt * 16 + (lane & 15);
  int k0 = kc * 32 + ((lane & 16) ? 8 : 0);
  const _Float16* p = W + m * ld + k0;
  v8h lo = *(const v8h*)p;
  v8h hi = *(const v8h*)(p + 16);
  v16h a;
#pragma unroll
  for (int i = 0; i < 8; ++i) { a[i] = lo[i]; a[i + 8] = hi[i]; }
  return a;
}

// B 32x16 f16 tile: lane holds column n = lane&15; elements 0..7 = k kbase..+7, 8..15 = k kbase+16..+23.
__device__ inline v16h load_b(const _Float16* row, int nt) {
  v16h b;
#pragma unroll
  for (int i = 0; i < 8; ++i) { b[i] = row[i * GS + nt * 16]; b[i + 8] = row[(i + 16) * GS + nt * 16]; }
  return b;
}

// ================= kernel 4: fused gather + 3-layer MLP + max ==========
// One wave per query. LDS staging per wave: G (96x32 padded, f16) and H
// (64x32 f16). D-layout f32 accumulators -> bias -> relu -> f16 -> LDS
// B-layout for the next layer. Final max over 32 samples via tile max +
// 4 shfl-xor steps. No cross-wave sharing -> no barriers needed.
__global__ __launch_bounds__(128) void mlp_kernel(
    const float* __restrict__ xyz, const float* __restrict__ feat,
    const float* __restrict__ new_xyz, const int* __restrict__ idxbuf,
    const _Float16* __restrict__ W1h, const _Float16* __restrict__ W2h,
    const _Float16* __restrict__ W3h, const float* __restrict__ b1,
    const float* __restrict__ b2, const float* __restrict__ b3,
    float* __restrict__ outf) {
  __shared__ __align__(16) _Float16 Gs[4][96 * GS];
  __shared__ __align__(16) _Float16 Hs[4][64 * GS];
  const int wid  = threadIdx.x >> 5;
  const int lane = threadIdx.x & 31;
  const int qg   = blockIdx.x * 4 + wid;
  const int b    = qg / NPOINT;
  const int q    = qg % NPOINT;
  _Float16* G = Gs[wid];
  _Float16* H = Hs[wid];

  // ---- gather grouped tile: rows = channels (0..2 centered xyz, 3..66 feat,
  // 67..95 zero pad), cols = samples; lane = sample ----
  const int   pid = idxbuf[((size_t)b * NPOINT + q) * NSAMPLE + lane];
  const float* cp = new_xyz + ((size_t)b * NPOINT + q) * 3;
  const float cx = cp[0], cy = cp[1], cz = cp[2];
  const float* pp = xyz + ((size_t)b * NPTS + pid) * 3;
  G[0 * GS + lane] = (_Float16)(pp[0] - cx);
  G[1 * GS + lane] = (_Float16)(pp[1] - cy);
  G[2 * GS + lane] = (_Float16)(pp[2] - cz);
  const float* fb = feat + (size_t)b * NCH * NPTS + pid;
#pragma unroll 4
  for (int c = 0; c < NCH; ++c)
    G[(3 + c) * GS + lane] = (_Float16)fb[(size_t)c * NPTS];
#pragma unroll
  for (int r = 67; r < 96; ++r) G[r * GS + lane] = (_Float16)0.f;

  const int moff = (lane & 16) ? 8 : 0;
  const int ncol = lane & 15;

  // ---- layer 1: H(64x32) = relu(W1(64x96p) * G(96x32) + b1) ----
  {
    v8f acc[4][2];
#pragma unroll
    for (int mt = 0; mt < 4; ++mt) { acc[mt][0] = {}; acc[mt][1] = {}; }
#pragma unroll
    for (int kc = 0; kc < 3; ++kc) {
      const _Float16* grow = G + (kc * 32 + ((lane & 16) ? 8 : 0)) * GS + (lane & 15);
      v16h bt0 = load_b(grow, 0);
      v16h bt1 = load_b(grow, 1);
#pragma unroll
      for (int mt = 0; mt < 4; ++mt) {
        v16h a = load_a(W1h, 96, mt, kc, lane);
        acc[mt][0] = wmma16(a, bt0, acc[mt][0]);
        acc[mt][1] = wmma16(a, bt1, acc[mt][1]);
      }
    }
#pragma unroll
    for (int mt = 0; mt < 4; ++mt) {
      const float* bp = b1 + mt * 16 + moff;
#pragma unroll
      for (int nt = 0; nt < 2; ++nt) {
#pragma unroll
        for (int r = 0; r < 8; ++r) {
          float h = acc[mt][nt][r] + bp[r];
          h = fmaxf(h, 0.f);
          H[(mt * 16 + moff + r) * GS + nt * 16 + ncol] = (_Float16)h;
        }
      }
    }
  }

  // ---- layer 2: G(rows 0..63) = relu(W2(64x64) * H(64x32) + b2) ----
  {
    v8f acc[4][2];
#pragma unroll
    for (int mt = 0; mt < 4; ++mt) { acc[mt][0] = {}; acc[mt][1] = {}; }
#pragma unroll
    for (int kc = 0; kc < 2; ++kc) {
      const _Float16* hrow = H + (kc * 32 + ((lane & 16) ? 8 : 0)) * GS + (lane & 15);
      v16h bt0 = load_b(hrow, 0);
      v16h bt1 = load_b(hrow, 1);
#pragma unroll
      for (int mt = 0; mt < 4; ++mt) {
        v16h a = load_a(W2h, 64, mt, kc, lane);
        acc[mt][0] = wmma16(a, bt0, acc[mt][0]);
        acc[mt][1] = wmma16(a, bt1, acc[mt][1]);
      }
    }
#pragma unroll
    for (int mt = 0; mt < 4; ++mt) {
      const float* bp = b2 + mt * 16 + moff;
#pragma unroll
      for (int nt = 0; nt < 2; ++nt) {
#pragma unroll
        for (int r = 0; r < 8; ++r) {
          float h = acc[mt][nt][r] + bp[r];
          h = fmaxf(h, 0.f);
          G[(mt * 16 + moff + r) * GS + nt * 16 + ncol] = (_Float16)h;
        }
      }
    }
  }

  // ---- layer 3: W3(128x64) * G2(64x32), max over 32 samples, +b3, relu ----
#pragma unroll
  for (int mt = 0; mt < 8; ++mt) {
    v8f a0 = {}, a1 = {};
#pragma unroll
    for (int kc = 0; kc < 2; ++kc) {
      const _Float16* grow = G + (kc * 32 + ((lane & 16) ? 8 : 0)) * GS + (lane & 15);
      v16h bt0 = load_b(grow, 0);
      v16h bt1 = load_b(grow, 1);
      v16h a = load_a(W3h, 64, mt, kc, lane);
      a0 = wmma16(a, bt0, a0);
      a1 = wmma16(a, bt1, a1);
    }
    float vmr[8];
#pragma unroll
    for (int r = 0; r < 8; ++r) vmr[r] = fmaxf(a0[r], a1[r]);  // n vs n+16
#pragma unroll
    for (int off = 1; off < 16; off <<= 1) {
#pragma unroll
      for (int r = 0; r < 8; ++r) vmr[r] = fmaxf(vmr[r], __shfl_xor(vmr[r], off));
    }
    if (ncol == 0) {  // lanes 0 and 16 hold row-halves mt*16+{0..7} / +{8..15}
#pragma unroll
      for (int r = 0; r < 8; ++r) {
        int m = mt * 16 + moff + r;
        float v = vmr[r] + b3[m];
        v = fmaxf(v, 0.f);
        outf[((size_t)b * 128 + m) * NPOINT + q] = v;
      }
    }
  }
}

// ================= launch ==================
extern "C" void kernel_launch(void* const* d_in, const int* in_sizes, int n_in,
                              void* d_out, int out_size, void* d_ws, size_t ws_size,
                              hipStream_t stream) {
  (void)in_sizes; (void)n_in; (void)out_size; (void)ws_size;
  const float* xyz  = (const float*)d_in[0];
  const float* feat = (const float*)d_in[1];
  const float* W1   = (const float*)d_in[2];
  const float* b1   = (const float*)d_in[3];
  const float* W2   = (const float*)d_in[4];
  const float* b2   = (const float*)d_in[5];
  const float* W3   = (const float*)d_in[6];
  const float* b3   = (const float*)d_in[7];

  float* new_xyz = (float*)d_out;                       // (8,1024,3)
  float* outf    = (float*)d_out + NB * NPOINT * 3;     // (8,128,1024)

  int* idxbuf = (int*)d_ws;                             // (8,1024,32) int
  _Float16* W1h = (_Float16*)((char*)d_ws + (size_t)NB * NPOINT * NSAMPLE * 4);
  _Float16* W2h = W1h + 64 * 96;
  _Float16* W3h = W2h + 64 * 64;

  prep_weights<<<64, 256, 0, stream>>>(W1, W2, W3, W1h, W2h, W3h);
  fps_kernel<<<NB, 1024, 0, stream>>>(xyz, new_xyz);
  ballq_kernel<<<(NB * NPOINT * 32) / 256, 256, 0, stream>>>(xyz, new_xyz, idxbuf);
  mlp_kernel<<<(NB * NPOINT) / 4, 128, 0, stream>>>(xyz, feat, new_xyz, idxbuf, W1h,
                                                    W2h, W3h, b1, b2, b3, outf);
}
